// Probsparse_Attention_Head_1752346657140
// MI455X (gfx1250) — matrix operand order repaired and checked by reading, not compiled, not run on hardware
//
#include <hip/hip_runtime.h>
#include <math.h>

typedef unsigned short u16;
typedef unsigned int   u32;
typedef __attribute__((ext_vector_type(16))) __bf16 v16bf;
typedef __attribute__((ext_vector_type(8)))  __bf16 v8bf;
typedef __attribute__((ext_vector_type(8)))  float  v8f;

// Fixed problem shape from setup_inputs()
constexpr int B   = 4;
constexpr int Q   = 2048;
constexpr int KV  = 2048;
constexpr int D   = 512;
constexpr int UTOP = 60;            // int(8 * ln(2048)) = 60 selected queries

// ---------------------------------------------------------------------------
// f32 -> bf16 (round-to-nearest-even), packed into workspace
// ---------------------------------------------------------------------------
__global__ void cvt_bf16_kernel(const float* __restrict__ src,
                                u16* __restrict__ dst, int n) {
  int i = blockIdx.x * blockDim.x + threadIdx.x;
  if (i < n) {
    u32 x = __float_as_uint(src[i]);
    u32 r = x + 0x7FFFu + ((x >> 16) & 1u);
    dst[i] = (u16)(r >> 16);
  }
}

// ---------------------------------------------------------------------------
// Per-batch histogram of sample_idx -> counts[b][kv]  (LDS atomics)
// ---------------------------------------------------------------------------
__global__ void hist_kernel(const int* __restrict__ sample_idx,
                            int* __restrict__ counts, int Usamp) {
  __shared__ int h[KV];
  int b = blockIdx.x;
  for (int i = threadIdx.x; i < KV; i += blockDim.x) h[i] = 0;
  __syncthreads();
  const int* s = sample_idx + (size_t)b * Usamp;
  for (int i = threadIdx.x; i < Usamp; i += blockDim.x) atomicAdd(&h[s[i]], 1);
  __syncthreads();
  for (int i = threadIdx.x; i < KV; i += blockDim.x) counts[b * KV + i] = h[i];
}

// ---------------------------------------------------------------------------
// Fused S = Q*K^T (bf16 WMMA, f32 accum) -> M[b,q] = masked_max - wsum/U.
// Block = 8 waves = 128 query rows; each wave owns 16 rows and keeps the full
// A-strip (16 x 512 bf16 = 16 fragments) in registers. K tiles (16 cols x 512)
// are staged in LDS once per block and re-read by all 8 waves.
// ---------------------------------------------------------------------------
constexpr int LKS = D + 8;          // padded LDS row stride (ushorts) for banks

__global__ __launch_bounds__(256)
void sparsity_kernel(const u16* __restrict__ qbf, const u16* __restrict__ kbf,
                     const int* __restrict__ counts, float* __restrict__ Mout,
                     float invU) {
  __shared__ __align__(16) u16 lk[16 * LKS];   // ~16.6 KB
  const int b     = blockIdx.x;
  const int wave  = threadIdx.x >> 5;          // 0..7 (wave32)
  const int lane  = threadIdx.x & 31;
  const int hi    = lane >> 4;                 // half-wave select
  const int m     = lane & 15;
  const int qbase = blockIdx.y * 128 + wave * 16;

  // Load A fragments: 16 chunks of K=32 (ISA 16-bit A layout: lane l%16 = row,
  // elems[0..7] = K hi*8.., elems[8..15] = K 16+hi*8..)
  v16bf afrag[16];
  const u16* qrow = qbf + ((size_t)b * Q + qbase + m) * D;
  #pragma unroll
  for (int t = 0; t < 16; ++t) {
    const int kb = t * 32;
    v8bf lo = *(const v8bf*)(qrow + kb + hi * 8);
    v8bf up = *(const v8bf*)(qrow + kb + 16 + hi * 8);
    #pragma unroll
    for (int e = 0; e < 8; ++e) { afrag[t][e] = lo[e]; afrag[t][8 + e] = up[e]; }
  }

  float wmax[8], wsum[8];
  #pragma unroll
  for (int r = 0; r < 8; ++r) { wmax[r] = -INFINITY; wsum[r] = 0.0f; }

  for (int nt = 0; nt < KV / 16; ++nt) {
    __syncthreads();                           // previous tile fully consumed
    {   // cooperative stage of 16 k-rows (columns of S) into padded LDS
      const u32* src = (const u32*)(kbf + ((size_t)b * KV + nt * 16) * D);
      u32* dst = (u32*)lk;
      #pragma unroll
      for (int i = 0; i < 16; ++i)             // 16 rows x 256 dwords
        dst[i * (LKS / 2) + threadIdx.x] = src[i * (D / 2) + threadIdx.x];
    }
    __syncthreads();

    v8f acc = {};
    const u16* kcol = lk + (size_t)m * LKS + hi * 8;    // B layout: lane=col
    #pragma unroll
    for (int t = 0; t < 16; ++t) {
      v8bf blo = *(const v8bf*)(kcol + t * 32);
      v8bf bup = *(const v8bf*)(kcol + t * 32 + 16);
      v16bf bfrag;
      #pragma unroll
      for (int e = 0; e < 8; ++e) { bfrag[e] = blo[e]; bfrag[8 + e] = bup[e]; }
      acc = __builtin_amdgcn_wmma_f32_16x16x32_bf16(
          false, afrag[t], false, bfrag, (short)0, acc, false, false);
    }

    // Fold this 16x16 S tile into M accumulators. C/D layout: VGPR r holds
    // row (hi*8 + r) at column N = m for this lane.
    const int cnt = counts[b * KV + nt * 16 + m];
    const float fc = (float)cnt;
    #pragma unroll
    for (int r = 0; r < 8; ++r) {
      float s = acc[r];
      wsum[r] += fc * s;                       // sum over sampled keys (w/ dup)
      if (cnt > 0) wmax[r] = fmaxf(wmax[r], s);// max over sampled key set
    }
  }

  // Reduce across the 16 columns held by lanes of the same half-wave.
  #pragma unroll
  for (int sft = 1; sft < 16; sft <<= 1) {
    #pragma unroll
    for (int r = 0; r < 8; ++r) {
      wmax[r] = fmaxf(wmax[r], __shfl_xor(wmax[r], sft, 32));
      wsum[r] += __shfl_xor(wsum[r], sft, 32);
    }
  }
  if (m == 0) {
    #pragma unroll
    for (int r = 0; r < 8; ++r) {
      int row = qbase + hi * 8 + r;
      Mout[b * Q + row] = wmax[r] - wsum[r] * invU;
    }
  }
}

// ---------------------------------------------------------------------------
// Iterative top-60 per batch over M (2048 values); ties -> lower index to
// match lax.top_k.
// ---------------------------------------------------------------------------
__global__ void topk_kernel(const float* __restrict__ Mvals,
                            int* __restrict__ topidx) {
  __shared__ float vals[Q];
  __shared__ float rv[256];
  __shared__ int   ri[256];
  int b = blockIdx.x, t = threadIdx.x;
  for (int i = t; i < Q; i += 256) vals[i] = Mvals[b * Q + i];
  __syncthreads();
  for (int sel = 0; sel < UTOP; ++sel) {
    float best = -INFINITY; int bi = 0x7fffffff;
    for (int i = t; i < Q; i += 256) {
      float v = vals[i];
      if (v > best || (v == best && i < bi)) { best = v; bi = i; }
    }
    rv[t] = best; ri[t] = bi;
    __syncthreads();
    for (int s = 128; s > 0; s >>= 1) {
      if (t < s) {
        float ov = rv[t + s]; int oi = ri[t + s];
        if (ov > rv[t] || (ov == rv[t] && oi < ri[t])) { rv[t] = ov; ri[t] = oi; }
      }
      __syncthreads();
    }
    if (t == 0) { topidx[b * UTOP + sel] = ri[0]; vals[ri[0]] = -INFINITY; }
    __syncthreads();
  }
}

// ---------------------------------------------------------------------------
// vmean[b][d] = mean over kv of v
// ---------------------------------------------------------------------------
__global__ void vmean_kernel(const float* __restrict__ v,
                             float* __restrict__ vmean) {
  int b = blockIdx.x, d = threadIdx.x;      // blockDim = 512
  const float* vb = v + (size_t)b * KV * D;
  float s = 0.0f;
  for (int kv = 0; kv < KV; ++kv) s += vb[(size_t)kv * D + d];
  vmean[b * D + d] = s * (1.0f / KV);
}

// out[b][q][d] = vmean[b][d]
__global__ void fill_kernel(const float* __restrict__ vmean,
                            float* __restrict__ out) {
  size_t i = (size_t)blockIdx.x * blockDim.x + threadIdx.x;
  if (i < (size_t)B * Q * D) {
    int d = (int)(i % D);
    int b = (int)(i / ((size_t)Q * D));
    out[i] = vmean[b * D + d];
  }
}

// ---------------------------------------------------------------------------
// Exact f32 attention for the 60 selected queries per batch; scatter into out.
// ---------------------------------------------------------------------------
__global__ __launch_bounds__(256)
void attn_kernel(const float* __restrict__ q, const float* __restrict__ k,
                 const float* __restrict__ v, const int* __restrict__ topidx,
                 float* __restrict__ out) {
  __shared__ float qrow[D];
  __shared__ float p[KV];
  __shared__ float red[256];
  const int b = blockIdx.x, j = blockIdx.y, t = threadIdx.x;
  const int qi = topidx[b * UTOP + j];

  const float* qp = q + ((size_t)b * Q + qi) * D;
  for (int i = t; i < D; i += 256) qrow[i] = qp[i];
  __syncthreads();

  const float* kb = k + (size_t)b * KV * D;
  const float scale = rsqrtf((float)KV);
  for (int kv = t; kv < KV; kv += 256) {
    const float* kr = kb + (size_t)kv * D;
    float acc = 0.0f;
    for (int d = 0; d < D; d += 4)
      acc += qrow[d] * kr[d] + qrow[d + 1] * kr[d + 1] +
             qrow[d + 2] * kr[d + 2] + qrow[d + 3] * kr[d + 3];
    p[kv] = acc * scale;
  }
  __syncthreads();

  float mx = -INFINITY;
  for (int kv = t; kv < KV; kv += 256) mx = fmaxf(mx, p[kv]);
  red[t] = mx; __syncthreads();
  for (int s = 128; s > 0; s >>= 1) {
    if (t < s) red[t] = fmaxf(red[t], red[t + s]);
    __syncthreads();
  }
  mx = red[0];
  __syncthreads();

  float sum = 0.0f;
  for (int kv = t; kv < KV; kv += 256) {
    float e = __expf(p[kv] - mx);
    p[kv] = e;
    sum += e;
  }
  red[t] = sum; __syncthreads();
  for (int s = 128; s > 0; s >>= 1) {
    if (t < s) red[t] += red[t + s];
    __syncthreads();
  }
  const float inv = 1.0f / red[0];

  const float* vb = v + (size_t)b * KV * D;
  float* orow = out + ((size_t)b * Q + qi) * D;
  for (int d = t; d < D; d += 256) {
    float acc = 0.0f;
    for (int kv = 0; kv < KV; ++kv) acc += p[kv] * vb[(size_t)kv * D + d];
    orow[d] = acc * inv;
  }
}

// ---------------------------------------------------------------------------
extern "C" void kernel_launch(void* const* d_in, const int* in_sizes, int n_in,
                              void* d_out, int out_size, void* d_ws, size_t ws_size,
                              hipStream_t stream) {
  const float* q    = (const float*)d_in[0];
  const float* k    = (const float*)d_in[1];
  const float* v    = (const float*)d_in[2];
  const int*   sidx = (const int*)d_in[3];
  float* out = (float*)d_out;
  const int Usamp = in_sizes[3] / B;           // 15615

  char* ws = (char*)d_ws;
  u16* qbf     = (u16*)ws;  ws += (size_t)B * Q * D * 2;
  u16* kbf     = (u16*)ws;  ws += (size_t)B * KV * D * 2;
  int* counts  = (int*)ws;  ws += (size_t)B * KV * 4;
  float* Mvals = (float*)ws; ws += (size_t)B * Q * 4;
  int* topidx  = (int*)ws;  ws += 1024;
  float* vmean = (float*)ws;

  const int nq = B * Q * D;
  cvt_bf16_kernel<<<(nq + 255) / 256, 256, 0, stream>>>(q, qbf, nq);
  cvt_bf16_kernel<<<(B * KV * D + 255) / 256, 256, 0, stream>>>(k, kbf, B * KV * D);
  hist_kernel<<<B, 1024, 0, stream>>>(sidx, counts, Usamp);

  dim3 gM(B, Q / 128);
  sparsity_kernel<<<gM, 256, 0, stream>>>(qbf, kbf, counts, Mvals,
                                          1.0f / (float)Usamp);

  vmean_kernel<<<B, D, 0, stream>>>(v, vmean);
  fill_kernel<<<(nq + 255) / 256, 256, 0, stream>>>(vmean, out);

  topk_kernel<<<B, 256, 0, stream>>>(Mvals, topidx);

  dim3 gA(B, UTOP);
  attn_kernel<<<gA, 256, 0, stream>>>(q, k, v, topidx, out);
}
